// SE_FBCNet_spatial_29360396435889
// MI455X (gfx1250) — matrix operand order repaired and checked by reading, not compiled, not run on hardware
//
#include <hip/hip_runtime.h>
#include <hip/hip_bf16.h>
#include <math.h>

typedef __attribute__((ext_vector_type(16))) __bf16 v16bf;
typedef __attribute__((ext_vector_type(8)))  float  v8f;

#define B_     64
#define NBANDS 9
#define NCHAN  65
#define NTIME  1000
#define NCLASS 4
#define SF     4
#define G_     5
#define W_     13
#define SM     2
#define NFEAT  90      // NBANDS*SM*G
#define NOUT   432     // M*NBANDS
#define NT16   63      // ceil(1000/16) N tiles
#define NK32   3       // ceil(90/32)  K tiles
#define NM16   27      // 432/16       M tiles

// ---------------------------------------------------------------------------
// K1: xin = x + loc ; grouped spatial conv with renormed weights (maxnorm 2)
// y[b,f,t] = sum_w cw[g,n,j,w] * (x[b,n,idx[g,w],t] + loc[idx[g,w],t]) + conv_b
// feature index f = (n*SM + j)*G + g  (matches feat reshape order)
// ---------------------------------------------------------------------------
__global__ void conv_kernel(const float* __restrict__ x,
                            const float* __restrict__ loc,
                            const int*   __restrict__ idx,
                            const float* __restrict__ conv_w,
                            const float* __restrict__ conv_b,
                            float* __restrict__ y)
{
    int f = blockIdx.y;                    // 0..89
    int b = blockIdx.z;                    // 0..63
    int t = blockIdx.x * 256 + threadIdx.x;
    int g = f % G_;
    int nj = f / G_;
    int j = nj % SM;
    int n = nj / SM;

    __shared__ float s_cw[W_];
    __shared__ int   s_ch[W_];
    if (threadIdx.x < W_) {
        s_cw[threadIdx.x] = conv_w[(((g * NBANDS) + n) * SM + j) * W_ + threadIdx.x];
        s_ch[threadIdx.x] = idx[g * W_ + threadIdx.x];
    }
    __syncthreads();

    float nrm2 = 0.f;
#pragma unroll
    for (int w = 0; w < W_; ++w) nrm2 += s_cw[w] * s_cw[w];
    float nrm = sqrtf(nrm2);
    float sc = (nrm > 2.0f) ? (2.0f / (nrm + 1e-7f)) : 1.0f;

    if (t >= NTIME) return;
    float acc = conv_b[(g * NBANDS + n) * SM + j];
#pragma unroll
    for (int w = 0; w < W_; ++w) {
        int ch = s_ch[w];
        float xv = x[(((size_t)b * NBANDS + n) * NCHAN + ch) * NTIME + t]
                 + loc[ch * NTIME + t];
        acc += sc * s_cw[w] * xv;
    }
    y[((size_t)b * NFEAT + f) * NTIME + t] = acc;
}

// ---------------------------------------------------------------------------
// K2: batch-norm statistics over (B, T) per feature; fold gamma/beta:
// stats[2f] = g*invstd, stats[2f+1] = b - mean*g*invstd
// ---------------------------------------------------------------------------
__global__ void bn_stats_kernel(const float* __restrict__ y,
                                const float* __restrict__ bn_g,
                                const float* __restrict__ bn_b,
                                float* __restrict__ stats)
{
    int f = blockIdx.x;
    int tid = threadIdx.x;
    float s = 0.f, sq = 0.f;
    for (int i = tid; i < B_ * NTIME; i += 256) {
        int b = i / NTIME, t = i % NTIME;
        float v = y[((size_t)b * NFEAT + f) * NTIME + t];
        s += v; sq += v * v;
    }
    __shared__ float rs[256], rq[256];
    rs[tid] = s; rq[tid] = sq;
    __syncthreads();
    for (int o = 128; o > 0; o >>= 1) {
        if (tid < o) { rs[tid] += rs[tid + o]; rq[tid] += rq[tid + o]; }
        __syncthreads();
    }
    if (tid == 0) {
        const float inv_n = 1.0f / (B_ * NTIME);
        float mean = rs[0] * inv_n;
        float var  = rq[0] * inv_n - mean * mean;
        float inv  = rsqrtf(var + 1e-5f);
        int g = f % G_, nj = f / G_, j = nj % SM, n = nj / SM;
        float gg = bn_g[(g * NBANDS + n) * SM + j];
        float bb = bn_b[(g * NBANDS + n) * SM + j];
        float sA = gg * inv;
        stats[2 * f]     = sA;
        stats[2 * f + 1] = bb - mean * sA;
    }
}

// ---------------------------------------------------------------------------
// K3a: pack lin_w (432x90) into WMMA A-fragment tiles (16-bit A 16x32 layout):
// lane l: M = l&15, half = l>>4 ; elem e: v=e/2,p=e&1 ->
//   K = v<4 ? 2v+p+8*half : 16+2(v-4)+p+8*half
// ---------------------------------------------------------------------------
__global__ void pack_a_kernel(const float* __restrict__ lin_w,
                              __bf16* __restrict__ Apack)
{
    int tile = blockIdx.x;      // 0 .. NM16*NK32-1
    int lane = threadIdx.x;     // 32 threads
    int mt = tile / NK32, kt = tile % NK32;
    int half = lane >> 4;
    int o = mt * 16 + (lane & 15);
    v16bf v;
#pragma unroll
    for (int e = 0; e < 16; ++e) {
        int vv = e >> 1, p = e & 1;
        int kL = (vv < 4) ? (2 * vv + p + 8 * half)
                          : (16 + 2 * (vv - 4) + p + 8 * half);
        int f = kt * 32 + kL;
        v[e] = (__bf16)((f < NFEAT) ? lin_w[o * NFEAT + f] : 0.f);
    }
    *(v16bf*)(Apack + ((size_t)tile * 512 + lane * 16)) = v;
}

// ---------------------------------------------------------------------------
// K3b: BN + SiLU + bf16 + pack activations into WMMA B-fragment tiles
// (16-bit B 32x16: lanes 0-15 K=0..7,16..23, lanes 16-31 K=8..15,24..31, N = lane&15)
// Bpack[((b*NT16 + nt)*NK32 + kt)] tile of 512 bf16, 32B/lane contiguous
// ---------------------------------------------------------------------------
__global__ void pack_b_kernel(const float* __restrict__ y,
                              const float* __restrict__ stats,
                              __bf16* __restrict__ Bpack)
{
    int slot = blockIdx.x * 8 + (threadIdx.x >> 5);
    int lane = threadIdx.x & 31;
    if (slot >= B_ * NT16 * NK32) return;
    int b  = slot / (NT16 * NK32);
    int r  = slot % (NT16 * NK32);
    int nt = r / NK32;
    int kt = r % NK32;
    int half = lane >> 4;
    int t = nt * 16 + (lane & 15);
    v16bf v;
#pragma unroll
    for (int e = 0; e < 16; ++e) {
        int f = kt * 32 + ((e < 8) ? (e + 8 * half) : (16 + 8 * half + (e - 8)));   // K = e<8 ? e+8*half : 16+8*half+(e-8)
        float val = 0.f;
        if (f < NFEAT && t < NTIME) {
            float raw = y[((size_t)b * NFEAT + f) * NTIME + t];
            float z = raw * stats[2 * f] + stats[2 * f + 1];
            val = z / (1.f + __expf(-z));  // SiLU
        }
        v[e] = (__bf16)val;
    }
    *(v16bf*)(Bpack + ((size_t)slot * 512 + lane * 16)) = v;
}

// ---------------------------------------------------------------------------
// K4: per (b, M-tile) workgroup: z = lin_w @ feat_b via v_wmma_f32_16x16x32_bf16,
// fused windowed variance (250 samples, ddof=1) -> log(clip(var)) in zlog.
// Wave w owns N-tiles [8w, 8w+8) (contiguous t-range of 128), so the window
// index s = t/250 changes at most once per lane across its loop:
// sum/sumsq accumulate in registers and flush to LDS only on s-change.
// B fragments are software-pipelined (next tile prefetched before WMMAs).
// ---------------------------------------------------------------------------
__global__ void gemm_var_kernel(const __bf16* __restrict__ Apack,
                                const __bf16* __restrict__ Bpack,
                                const float* __restrict__ lin_b,
                                float* __restrict__ zlog)
{
    int mt  = blockIdx.x;           // 0..26
    int b   = blockIdx.y;           // 0..63
    int tid = threadIdx.x;
    int wave = tid >> 5, lane = tid & 31;

    __shared__ float lsum[16 * SF], lsq[16 * SF];
    if (tid < 64) { lsum[tid] = 0.f; lsq[tid] = 0.f; }
    __syncthreads();

    // A fragments for this M-tile (reused across all N tiles)
    v16bf a0 = *(const v16bf*)(Apack + ((size_t)(mt * NK32 + 0) * 512 + lane * 16));
    v16bf a1 = *(const v16bf*)(Apack + ((size_t)(mt * NK32 + 1) * 512 + lane * 16));
    v16bf a2 = *(const v16bf*)(Apack + ((size_t)(mt * NK32 + 2) * 512 + lane * 16));

    int halfm = lane >> 4, nloc = lane & 15;

    // bias values for this lane's 8 accumulator rows (hoisted out of the loop)
    float lb[8];
#pragma unroll
    for (int rr = 0; rr < 8; ++rr) lb[rr] = lin_b[mt * 16 + rr + 8 * halfm];

    int nt0 = wave * 8;
    int ntiles = (NT16 - nt0 < 8) ? (NT16 - nt0) : 8;   // wave 7: 7 tiles

    // per-lane register accumulators for the current variance window
    float asum[8], asq[8];
#pragma unroll
    for (int rr = 0; rr < 8; ++rr) { asum[rr] = 0.f; asq[rr] = 0.f; }
    int cur_s = -1;

    // prime the software pipeline
    const __bf16* bp = Bpack + ((size_t)(b * NT16 + nt0) * NK32) * 512 + lane * 16;
    v16bf nb0 = *(const v16bf*)(bp);
    v16bf nb1 = *(const v16bf*)(bp + 512);
    v16bf nb2 = *(const v16bf*)(bp + 1024);

    for (int i = 0; i < ntiles; ++i) {
        v16bf b0 = nb0, b1 = nb1, b2 = nb2;
        if (i + 1 < ntiles) {                    // prefetch next tile's fragments
            bp += 1536;
            nb0 = *(const v16bf*)(bp);
            nb1 = *(const v16bf*)(bp + 512);
            nb2 = *(const v16bf*)(bp + 1024);
        }
        v8f c = {};
        c = __builtin_amdgcn_wmma_f32_16x16x32_bf16(false, a0, false, b0, (short)0, c, false, false);
        c = __builtin_amdgcn_wmma_f32_16x16x32_bf16(false, a1, false, b1, (short)0, c, false, false);
        c = __builtin_amdgcn_wmma_f32_16x16x32_bf16(false, a2, false, b2, (short)0, c, false, false);

        int t = (nt0 + i) * 16 + nloc;
        if (t < NTIME) {
            int s = t / 250;
            if (s != cur_s) {                    // at most once per lane (+ first)
                if (cur_s >= 0) {
#pragma unroll
                    for (int rr = 0; rr < 8; ++rr) {
                        int mloc = rr + 8 * halfm;
                        atomicAdd(&lsum[mloc * SF + cur_s], asum[rr]);
                        atomicAdd(&lsq [mloc * SF + cur_s], asq[rr]);
                        asum[rr] = 0.f; asq[rr] = 0.f;
                    }
                }
                cur_s = s;
            }
#pragma unroll
            for (int rr = 0; rr < 8; ++rr) {     // C layout: M = rr + 8*(lane>>4)
                float zv = c[rr] + lb[rr];
                asum[rr] += zv;
                asq[rr]  += zv * zv;
            }
        }
    }
    if (cur_s >= 0) {                            // final flush
#pragma unroll
        for (int rr = 0; rr < 8; ++rr) {
            int mloc = rr + 8 * halfm;
            atomicAdd(&lsum[mloc * SF + cur_s], asum[rr]);
            atomicAdd(&lsq [mloc * SF + cur_s], asq[rr]);
        }
    }
    __syncthreads();
    if (tid < 64) {
        int mloc = tid >> 2, s = tid & 3;
        float sm  = lsum[tid];
        float var = (lsq[tid] - sm * sm * (1.f / 250.f)) * (1.f / 249.f);
        var = fminf(fmaxf(var, 1e-6f), 1e6f);
        zlog[((size_t)b * NOUT + mt * 16 + mloc) * SF + s] = logf(var);
        (void)s;
    }
}

// ---------------------------------------------------------------------------
// K5: per-channel 4x4 attention, renormed classifier (maxnorm 0.5), log_softmax
// one block per batch element
// ---------------------------------------------------------------------------
__global__ void head_kernel(const float* __restrict__ zlog,
                            const float* __restrict__ aiw,
                            const float* __restrict__ aib,
                            const float* __restrict__ aow,
                            const float* __restrict__ aob,
                            const float* __restrict__ last_w,
                            const float* __restrict__ last_b,
                            float* __restrict__ out)
{
    int b = blockIdx.x;
    int tid = threadIdx.x;
    __shared__ float flatv[NOUT * SF];          // 1728 floats

    for (int c = tid; c < NOUT; c += 256) {
        float z[SF], q[SF], k[SF], v[SF];
#pragma unroll
        for (int i = 0; i < SF; ++i) z[i] = zlog[((size_t)b * NOUT + c) * SF + i];
        float wq = aiw[c * 3 + 0], bq = aib[c * 3 + 0];
        float wk = aiw[c * 3 + 1], bk = aib[c * 3 + 1];
        float wv = aiw[c * 3 + 2], bv = aib[c * 3 + 2];
#pragma unroll
        for (int i = 0; i < SF; ++i) {
            q[i] = z[i] * wq + bq;
            k[i] = z[i] * wk + bk;
            v[i] = z[i] * wv + bv;
        }
#pragma unroll
        for (int i = 0; i < SF; ++i) {
            float mx = q[i] * k[0];
#pragma unroll
            for (int j = 1; j < SF; ++j) mx = fmaxf(mx, q[i] * k[j]);
            float den = 0.f, ov = 0.f;
#pragma unroll
            for (int j = 0; j < SF; ++j) {
                float e = __expf(q[i] * k[j] - mx);
                den += e; ov += e * v[j];
            }
            flatv[c * SF + i] = (ov / den) * aow[c] + aob[c];
        }
    }
    __syncthreads();

    float dot[NCLASS] = {0.f, 0.f, 0.f, 0.f};
    float sqr[NCLASS] = {0.f, 0.f, 0.f, 0.f};
    for (int j = tid; j < NOUT * SF; j += 256) {
        float fv = flatv[j];
#pragma unroll
        for (int r = 0; r < NCLASS; ++r) {
            float w = last_w[r * (NOUT * SF) + j];
            dot[r] += w * fv;
            sqr[r] += w * w;
        }
    }
    __shared__ float rd[NCLASS][256], rq[NCLASS][256];
#pragma unroll
    for (int r = 0; r < NCLASS; ++r) { rd[r][tid] = dot[r]; rq[r][tid] = sqr[r]; }
    __syncthreads();
    for (int o = 128; o > 0; o >>= 1) {
        if (tid < o) {
#pragma unroll
            for (int r = 0; r < NCLASS; ++r) {
                rd[r][tid] += rd[r][tid + o];
                rq[r][tid] += rq[r][tid + o];
            }
        }
        __syncthreads();
    }
    if (tid == 0) {
        float logits[NCLASS];
#pragma unroll
        for (int r = 0; r < NCLASS; ++r) {
            float n  = sqrtf(rq[r][0]);
            float sc = (n > 0.5f) ? (0.5f / (n + 1e-7f)) : 1.0f;
            logits[r] = rd[r][0] * sc + last_b[r];
        }
        float mx = logits[0];
#pragma unroll
        for (int r = 1; r < NCLASS; ++r) mx = fmaxf(mx, logits[r]);
        float lse = 0.f;
#pragma unroll
        for (int r = 0; r < NCLASS; ++r) lse += __expf(logits[r] - mx);
        lse = logf(lse);
#pragma unroll
        for (int r = 0; r < NCLASS; ++r) out[b * NCLASS + r] = logits[r] - mx - lse;
    }
}

// ---------------------------------------------------------------------------
extern "C" void kernel_launch(void* const* d_in, const int* in_sizes, int n_in,
                              void* d_out, int out_size, void* d_ws, size_t ws_size,
                              hipStream_t stream)
{
    (void)in_sizes; (void)n_in; (void)out_size; (void)ws_size;
    const float* x      = (const float*)d_in[0];
    const float* loc    = (const float*)d_in[1];
    const int*   idx    = (const int*)  d_in[2];
    const float* conv_w = (const float*)d_in[3];
    const float* conv_b = (const float*)d_in[4];
    const float* bn_g   = (const float*)d_in[5];
    const float* bn_b   = (const float*)d_in[6];
    const float* lin_w  = (const float*)d_in[7];
    const float* lin_b  = (const float*)d_in[8];
    const float* aiw    = (const float*)d_in[9];
    const float* aib    = (const float*)d_in[10];
    const float* aow    = (const float*)d_in[11];
    const float* aob    = (const float*)d_in[12];
    const float* last_w = (const float*)d_in[13];
    const float* last_b = (const float*)d_in[14];
    float* out = (float*)d_out;

    char* ws = (char*)d_ws;
    size_t off = 0;
    auto wsalloc = [&](size_t bytes) -> void* {
        void* p = ws + off;
        off = (off + bytes + 255) & ~(size_t)255;
        return p;
    };
    float*  y     = (float*) wsalloc((size_t)B_ * NFEAT * NTIME * sizeof(float)); // 23.0 MB
    float*  stats = (float*) wsalloc(2 * NFEAT * sizeof(float));
    __bf16* Apack = (__bf16*)wsalloc((size_t)NM16 * NK32 * 512 * sizeof(__bf16)); // 83 KB
    __bf16* Bpack = (__bf16*)wsalloc((size_t)B_ * NT16 * NK32 * 512 * sizeof(__bf16)); // 12.4 MB
    float*  zlog  = (float*) wsalloc((size_t)B_ * NOUT * SF * sizeof(float));     // 0.44 MB

    conv_kernel<<<dim3(4, NFEAT, B_), 256, 0, stream>>>(x, loc, idx, conv_w, conv_b, y);
    bn_stats_kernel<<<NFEAT, 256, 0, stream>>>(y, bn_g, bn_b, stats);
    pack_a_kernel<<<NM16 * NK32, 32, 0, stream>>>(lin_w, Apack);
    int nslots = B_ * NT16 * NK32;
    pack_b_kernel<<<(nslots + 7) / 8, 256, 0, stream>>>(y, stats, Bpack);
    gemm_var_kernel<<<dim3(NM16, B_), 256, 0, stream>>>(Apack, Bpack, lin_b, zlog);
    head_kernel<<<B_, 256, 0, stream>>>(zlog, aiw, aib, aow, aob, last_w, last_b, out);
}
